// MultiHeadAttention_62809601737179
// MI455X (gfx1250) — matrix operand order repaired and checked by reading, not compiled, not run on hardware
//
#include <hip/hip_runtime.h>

typedef __attribute__((ext_vector_type(16))) __bf16 v16bf;
typedef __attribute__((ext_vector_type(8)))  __bf16 v8bf;
typedef __attribute__((ext_vector_type(8)))  float  v8f;

#define B_   4
#define S_   2048
#define D_   1024
#define H_   16
#define DEP  64

#ifndef __has_builtin
#define __has_builtin(x) 0
#endif

#if __has_builtin(__builtin_amdgcn_global_load_async_to_lds_b128)
#define HAVE_ASYNC_LDS 1
#else
#define HAVE_ASYNC_LDS 0
#endif

#if HAVE_ASYNC_LDS
typedef int b128_t __attribute__((vector_size(16)));
typedef __attribute__((address_space(1))) b128_t* gb128p;
typedef __attribute__((address_space(3))) b128_t* lb128p;
#endif

// Stage one 16-byte chunk global -> LDS (ASYNCcnt-tracked DMA when available).
__device__ __forceinline__ void stage16(void* ldst, const void* gsrc) {
#if HAVE_ASYNC_LDS
  __builtin_amdgcn_global_load_async_to_lds_b128(
      (gb128p)(uintptr_t)gsrc, (lb128p)(uintptr_t)ldst, 0, 0);
#else
  *(v8bf*)ldst = *(const v8bf*)gsrc;
#endif
}

// Wait until at most N of this wave's async LDS copies remain outstanding.
// Async loads complete in order, so N = "loads issued for the next tile"
// guarantees the current tile is resident while next-tile DMA stays in flight.
template <int N>
__device__ __forceinline__ void stage_wait() {
#if HAVE_ASYNC_LDS
#if __has_builtin(__builtin_amdgcn_s_wait_asynccnt)
  __builtin_amdgcn_s_wait_asynccnt(N);
#else
  asm volatile("s_wait_asynccnt %0" ::"i"(N) : "memory");
#endif
#endif
}

// float -> bf16, round-to-nearest-even
__device__ __forceinline__ __bf16 f2bf(float f) {
  unsigned int u = __builtin_bit_cast(unsigned int, f);
  unsigned int r = u + 0x7FFFu + ((u >> 16) & 1u);
  unsigned short hs = (unsigned short)(r >> 16);
  return __builtin_bit_cast(__bf16, hs);
}

// Build a 16x32 bf16 A-fragment from a row-major row pointer (stride-32 tile).
// Per-lane K = {half*16 + 0..15} (contiguous, matching B fragments).
__device__ __forceinline__ v16bf load_afrag(const __bf16* row, int half) {
  v8bf lo = *(const v8bf*)(row + half * 16);
  v8bf hi = *(const v8bf*)(row + half * 16 + 8);
  v16bf a;
#pragma unroll
  for (int i = 0; i < 8; ++i) { a[i] = lo[i]; a[i + 8] = hi[i]; }
  return a;
}

// ---------------------------------------------------------------------------
// f32 -> bf16 bulk conversion (8 elements / thread)
// ---------------------------------------------------------------------------
__global__ __launch_bounds__(256) void cvt_kernel(const float* __restrict__ src,
                                                  __bf16* __restrict__ dst, int n) {
  int i = (blockIdx.x * 256 + threadIdx.x) * 8;
  if (i + 8 <= n) {
    float4 a = *(const float4*)(src + i);
    float4 b = *(const float4*)(src + i + 4);
    v8bf o;
    o[0] = f2bf(a.x); o[1] = f2bf(a.y); o[2] = f2bf(a.z); o[3] = f2bf(a.w);
    o[4] = f2bf(b.x); o[5] = f2bf(b.y); o[6] = f2bf(b.z); o[7] = f2bf(b.w);
    *(v8bf*)(dst + i) = o;
  }
}

// ---------------------------------------------------------------------------
// Weight transpose + conversion: W[k][n] f32 -> WT[n][k] bf16.
// ---------------------------------------------------------------------------
__global__ __launch_bounds__(256) void wt_kernel(const float* __restrict__ W,
                                                 __bf16* __restrict__ WT) {
  const int idx = blockIdx.x * 256 + threadIdx.x;  // 0 .. 131071
  const int n  = idx & (D_ - 1);
  const int k0 = (idx >> 10) * 8;
  v8bf o;
#pragma unroll
  for (int i = 0; i < 8; ++i) o[i] = f2bf(W[(k0 + i) * D_ + n]);
  *(v8bf*)&WT[n * D_ + k0] = o;
}

// ---------------------------------------------------------------------------
// QKV GEMM: A[8192,1024]bf16 x WT[n][k]bf16 -> Q,K head-major [B,H,S,64],
// V transposed [B,H,64,S].  128x128 tile/WG, 8 waves x (16x128).
// Double-buffered async LDS staging.
// ---------------------------------------------------------------------------
__global__ __launch_bounds__(256) void gemm_qkv_kernel(
    const __bf16* __restrict__ A,
    const __bf16* __restrict__ WqT, const __bf16* __restrict__ WkT,
    const __bf16* __restrict__ WvT,
    __bf16* __restrict__ Qo, __bf16* __restrict__ Ko, __bf16* __restrict__ Vt) {
  __shared__ __attribute__((aligned(64))) __bf16 sA[2][128 * 32];   // [m][k]
  __shared__ __attribute__((aligned(64))) __bf16 sBT[2][128 * 32];  // [n][k]

  const int K = D_;
  const int m0 = blockIdx.x * 128;
  const int n0 = blockIdx.y * 128;
  const int z  = blockIdx.z;
  const __bf16* Bm = (z == 0) ? WqT : (z == 1) ? WkT : WvT;

  const int tid  = threadIdx.x;
  const int w    = tid >> 5;
  const int lane = tid & 31;
  const int half = (lane >> 4) & 1;
  const int l15  = lane & 15;

  // Staging coords: both tiles are [128 rows][32 k] = 512 16B chunks,
  // 2 chunks per thread per tile (4 async issues / thread / tile).
  const int r0 = tid >> 2,         kc0 = (tid & 3) * 8;
  const int r1 = (tid + 256) >> 2, kc1 = ((tid + 256) & 3) * 8;

#define QKV_STAGE(buf, kk)                                                  \
  do {                                                                      \
    stage16(&sA[buf][r0 * 32 + kc0],  &A[(m0 + r0) * K + (kk) + kc0]);      \
    stage16(&sBT[buf][r0 * 32 + kc0], &Bm[(n0 + r0) * K + (kk) + kc0]);     \
    stage16(&sA[buf][r1 * 32 + kc1],  &A[(m0 + r1) * K + (kk) + kc1]);      \
    stage16(&sBT[buf][r1 * 32 + kc1], &Bm[(n0 + r1) * K + (kk) + kc1]);     \
  } while (0)

  v8f acc[8] = {};

  QKV_STAGE(0, 0);
  for (int k0 = 0; k0 < K; k0 += 32) {
    const int cur = (k0 >> 5) & 1;
    if (k0 + 32 < K) {
      QKV_STAGE(cur ^ 1, k0 + 32);
      stage_wait<4>();   // current tile resident; next tile DMA in flight
    } else {
      stage_wait<0>();
    }
    __syncthreads();

    const v16bf af = load_afrag(&sA[cur][(w * 16 + l15) * 32], half);
    v16bf bfr[8];
#pragma unroll
    for (int j = 0; j < 8; ++j)
      bfr[j] = *(const v16bf*)&sBT[cur][(j * 16 + l15) * 32 + half * 16];
#pragma unroll
    for (int j = 0; j < 8; ++j)
      acc[j] = __builtin_amdgcn_wmma_f32_16x16x32_bf16(
          false, af, false, bfr[j], (short)0, acc[j], false, false);
    __syncthreads();   // readers done before this buffer is overwritten
  }
#undef QKV_STAGE

  // Epilogue: C layout row = r + 8*half, col = j*16 + l15.
#pragma unroll
  for (int j = 0; j < 8; ++j) {
    const int gcol = n0 + j * 16 + l15;
    const int hh = gcol >> 6, dep = gcol & 63;
#pragma unroll
    for (int r = 0; r < 8; ++r) {
      const int grow = m0 + w * 16 + r + half * 8;
      const int bb = grow >> 11, ss = grow & 2047;
      const __bf16 v = f2bf(acc[j][r]);
      if (z == 2) {
        Vt[(((bb * H_ + hh) * DEP) + dep) * S_ + ss] = v;      // [B,H,DEP,S]
      } else {
        __bf16* O = (z == 0) ? Qo : Ko;
        O[(((bb * H_ + hh) * S_) + ss) * DEP + dep] = v;       // [B,H,S,DEP]
      }
    }
  }
}

// ---------------------------------------------------------------------------
// Flash attention: one wave per 16-row query tile, 8 waves/WG share K/V tiles.
// Double-buffered async K/V staging; multiplicative mask; 1/sqrt(D)=1/32.
// ---------------------------------------------------------------------------
__global__ __launch_bounds__(256) void attn_kernel(
    const __bf16* __restrict__ Qb, const __bf16* __restrict__ Kb,
    const __bf16* __restrict__ Vt, const int* __restrict__ mask,
    __bf16* __restrict__ Ao) {
  __shared__ __attribute__((aligned(64))) __bf16 sK[2][32 * 64];    // [key][dep]
  __shared__ __attribute__((aligned(64))) __bf16 sVT[2][64 * 32];   // [dep][key]
  __shared__ __attribute__((aligned(64))) __bf16 sP[8][16 * 32];    // per-wave P
  __shared__ float sM[2][32];

  const int tid  = threadIdx.x;
  const int w    = tid >> 5;
  const int lane = tid & 31;
  const int half = (lane >> 4) & 1;
  const int l15  = lane & 15;
  const int h = blockIdx.y, b = blockIdx.z;
  const int q0 = (blockIdx.x * 8 + w) * 16;
  const int headbase  = ((b * H_ + h) * S_) * DEP;  // for Q, K
  const int vheadbase = ((b * H_ + h) * DEP) * S_;  // for Vt

  // Q fragments: 16x64 split into two 16x32 A-fragments
  const __bf16* qrow = Qb + headbase + (q0 + l15) * DEP;
  const v16bf qa0 = load_afrag(qrow, half);
  const v16bf qa1 = load_afrag(qrow + 32, half);

  float mrow[8], lrow[8];
#pragma unroll
  for (int r = 0; r < 8; ++r) { mrow[r] = -3.0e38f; lrow[r] = 0.f; }
  v8f accO[4] = {};

  const int kr = tid >> 3, kcg = (tid & 7) * 8;   // K tile: 32 x 64
  const int vd = tid >> 2, vcg = (tid & 3) * 8;   // Vt tile: 64 x 32

#define ATTN_STAGE(buf, kk)                                                   \
  do {                                                                        \
    stage16(&sK[buf][kr * 64 + kcg], &Kb[headbase + ((kk) + kr) * DEP + kcg]);\
    stage16(&sVT[buf][vd * 32 + vcg], &Vt[vheadbase + vd * S_ + (kk) + vcg]); \
    if (tid < 32)                                                             \
      sM[buf][tid] = (mask[b * S_ + (kk) + tid] == 0) ? -1.0e10f : 1.0f;      \
  } while (0)

  ATTN_STAGE(0, 0);
  for (int kt = 0; kt < S_; kt += 32) {
    const int cur = (kt >> 5) & 1;
    if (kt + 32 < S_) {
      ATTN_STAGE(cur ^ 1, kt + 32);
      stage_wait<2>();
    } else {
      stage_wait<0>();
    }
    __syncthreads();

    // scores: two 16-key sub-tiles, K-dim = 64 (two WMMA k-steps each)
    v8f s0 = {}, s1 = {};
    {
      const __bf16* kb = sK[cur];
      v16bf bk00 = *(const v16bf*)&kb[l15 * 64 + half * 16];
      v16bf bk01 = *(const v16bf*)&kb[l15 * 64 + 32 + half * 16];
      v16bf bk10 = *(const v16bf*)&kb[(16 + l15) * 64 + half * 16];
      v16bf bk11 = *(const v16bf*)&kb[(16 + l15) * 64 + 32 + half * 16];
      s0 = __builtin_amdgcn_wmma_f32_16x16x32_bf16(false, qa0, false, bk00,
                                                   (short)0, s0, false, false);
      s0 = __builtin_amdgcn_wmma_f32_16x16x32_bf16(false, qa1, false, bk01,
                                                   (short)0, s0, false, false);
      s1 = __builtin_amdgcn_wmma_f32_16x16x32_bf16(false, qa0, false, bk10,
                                                   (short)0, s1, false, false);
      s1 = __builtin_amdgcn_wmma_f32_16x16x32_bf16(false, qa1, false, bk11,
                                                   (short)0, s1, false, false);
    }
    // scale (1/32) and multiplicative mask, per key column
    const float mul0 = 0.03125f * sM[cur][l15];
    const float mul1 = 0.03125f * sM[cur][16 + l15];

    __bf16* pw = sP[w];
#pragma unroll
    for (int r = 0; r < 8; ++r) {
      float v0 = s0[r] * mul0;
      float v1 = s1[r] * mul1;
      float mx = fmaxf(v0, v1);
      for (int off = 1; off < 16; off <<= 1)
        mx = fmaxf(mx, __shfl_xor(mx, off, 32));
      const float mn   = fmaxf(mrow[r], mx);
      const float corr = __expf(mrow[r] - mn);
      const float p0   = __expf(v0 - mn);
      const float p1   = __expf(v1 - mn);
      float sum = p0 + p1;
      for (int off = 1; off < 16; off <<= 1) sum += __shfl_xor(sum, off, 32);
      lrow[r] = lrow[r] * corr + sum;
      mrow[r] = mn;
#pragma unroll
      for (int j = 0; j < 4; ++j) accO[j][r] *= corr;
      const int row = r + 8 * half;
      pw[row * 32 + l15]      = f2bf(p0);
      pw[row * 32 + 16 + l15] = f2bf(p1);
    }

    // P (16x32) x V (32x64): reload P as an A-fragment from per-wave LDS
    const v16bf pa = load_afrag(&pw[l15 * 32], half);
    v16bf vfr[4];
#pragma unroll
    for (int j = 0; j < 4; ++j)
      vfr[j] = *(const v16bf*)&sVT[cur][(j * 16 + l15) * 32 + half * 16];
#pragma unroll
    for (int j = 0; j < 4; ++j)
      accO[j] = __builtin_amdgcn_wmma_f32_16x16x32_bf16(
          false, pa, false, vfr[j], (short)0, accO[j], false, false);
    __syncthreads();
  }
#undef ATTN_STAGE

  // normalize and write merged-head [B,S,D] bf16
#pragma unroll
  for (int j = 0; j < 4; ++j) {
    const int col = h * DEP + j * 16 + l15;
#pragma unroll
    for (int r = 0; r < 8; ++r) {
      const int srow = q0 + r + 8 * half;
      Ao[(b * S_ + srow) * D_ + col] = f2bf(accO[j][r] / lrow[r]);
    }
  }
}

// ---------------------------------------------------------------------------
// Output projection GEMM (B = WoT[n][k]) + bias + inference BatchNorm, f32 out
// ---------------------------------------------------------------------------
__global__ __launch_bounds__(256) void gemm_out_kernel(
    const __bf16* __restrict__ A, const __bf16* __restrict__ BmT,
    const float* __restrict__ b_o, const float* __restrict__ gamma,
    const float* __restrict__ beta, const float* __restrict__ mmean,
    const float* __restrict__ mvar, float* __restrict__ out) {
  __shared__ __attribute__((aligned(64))) __bf16 sA[2][128 * 32];
  __shared__ __attribute__((aligned(64))) __bf16 sBT[2][128 * 32];

  const int K = D_;
  const int m0 = blockIdx.x * 128;
  const int n0 = blockIdx.y * 128;

  const int tid  = threadIdx.x;
  const int w    = tid >> 5;
  const int lane = tid & 31;
  const int half = (lane >> 4) & 1;
  const int l15  = lane & 15;

  const int r0 = tid >> 2,         kc0 = (tid & 3) * 8;
  const int r1 = (tid + 256) >> 2, kc1 = ((tid + 256) & 3) * 8;

#define OUT_STAGE(buf, kk)                                                  \
  do {                                                                      \
    stage16(&sA[buf][r0 * 32 + kc0],  &A[(m0 + r0) * K + (kk) + kc0]);      \
    stage16(&sBT[buf][r0 * 32 + kc0], &BmT[(n0 + r0) * K + (kk) + kc0]);    \
    stage16(&sA[buf][r1 * 32 + kc1],  &A[(m0 + r1) * K + (kk) + kc1]);      \
    stage16(&sBT[buf][r1 * 32 + kc1], &BmT[(n0 + r1) * K + (kk) + kc1]);    \
  } while (0)

  v8f acc[8] = {};

  OUT_STAGE(0, 0);
  for (int k0 = 0; k0 < K; k0 += 32) {
    const int cur = (k0 >> 5) & 1;
    if (k0 + 32 < K) {
      OUT_STAGE(cur ^ 1, k0 + 32);
      stage_wait<4>();
    } else {
      stage_wait<0>();
    }
    __syncthreads();

    const v16bf af = load_afrag(&sA[cur][(w * 16 + l15) * 32], half);
    v16bf bfr[8];
#pragma unroll
    for (int j = 0; j < 8; ++j)
      bfr[j] = *(const v16bf*)&sBT[cur][(j * 16 + l15) * 32 + half * 16];
#pragma unroll
    for (int j = 0; j < 8; ++j)
      acc[j] = __builtin_amdgcn_wmma_f32_16x16x32_bf16(
          false, af, false, bfr[j], (short)0, acc[j], false, false);
    __syncthreads();
  }
#undef OUT_STAGE

#pragma unroll
  for (int j = 0; j < 8; ++j) {
    const int gcol = n0 + j * 16 + l15;
    const float sc = gamma[gcol] * rsqrtf(mvar[gcol] + 1e-3f);
    const float sh = (b_o[gcol] - mmean[gcol]) * sc + beta[gcol];
#pragma unroll
    for (int r = 0; r < 8; ++r) {
      const int grow = m0 + w * 16 + r + half * 8;
      out[grow * D_ + gcol] = acc[j][r] * sc + sh;
    }
  }
}

// ---------------------------------------------------------------------------
extern "C" void kernel_launch(void* const* d_in, const int* in_sizes, int n_in,
                              void* d_out, int out_size, void* d_ws,
                              size_t ws_size, hipStream_t stream) {
  (void)in_sizes; (void)n_in; (void)out_size; (void)ws_size;
  const float* x     = (const float*)d_in[0];
  const int*   mask  = (const int*)d_in[1];
  const float* Wq    = (const float*)d_in[2];
  const float* Wk    = (const float*)d_in[3];
  const float* Wv    = (const float*)d_in[4];
  const float* Wo    = (const float*)d_in[5];
  const float* b_o   = (const float*)d_in[6];
  const float* gamma = (const float*)d_in[7];
  const float* beta  = (const float*)d_in[8];
  const float* mmean = (const float*)d_in[9];
  const float* mvar  = (const float*)d_in[10];
  float* out = (float*)d_out;

  char* ws = (char*)d_ws;
  size_t off = 0;
  auto alloc = [&](size_t bytes) -> void* {
    void* p = ws + off;
    off += (bytes + 255) & ~(size_t)255;
    return p;
  };
  __bf16* xb   = (__bf16*)alloc((size_t)B_ * S_ * D_ * 2);
  __bf16* WqT  = (__bf16*)alloc((size_t)D_ * D_ * 2);
  __bf16* WkT  = (__bf16*)alloc((size_t)D_ * D_ * 2);
  __bf16* WvT  = (__bf16*)alloc((size_t)D_ * D_ * 2);
  __bf16* WoT  = (__bf16*)alloc((size_t)D_ * D_ * 2);
  __bf16* Qb   = (__bf16*)alloc((size_t)B_ * S_ * D_ * 2);
  __bf16* Kb   = (__bf16*)alloc((size_t)B_ * S_ * D_ * 2);
  __bf16* Vt   = (__bf16*)alloc((size_t)B_ * S_ * D_ * 2);
  __bf16* Ab   = (__bf16*)alloc((size_t)B_ * S_ * D_ * 2);

  cvt_kernel<<<dim3((B_ * S_ * D_) / 2048), 256, 0, stream>>>(x, xb, B_ * S_ * D_);
  wt_kernel<<<dim3((D_ * D_) / (8 * 256)), 256, 0, stream>>>(Wq, WqT);
  wt_kernel<<<dim3((D_ * D_) / (8 * 256)), 256, 0, stream>>>(Wk, WkT);
  wt_kernel<<<dim3((D_ * D_) / (8 * 256)), 256, 0, stream>>>(Wv, WvT);
  wt_kernel<<<dim3((D_ * D_) / (8 * 256)), 256, 0, stream>>>(Wo, WoT);

  gemm_qkv_kernel<<<dim3(64, 8, 3), 256, 0, stream>>>(xb, WqT, WkT, WvT, Qb, Kb, Vt);

  attn_kernel<<<dim3(S_ / 16 / 8, H_, B_), 256, 0, stream>>>(Qb, Kb, Vt, mask, Ab);

  gemm_out_kernel<<<dim3(64, 8, 1), 256, 0, stream>>>(Ab, WoT, b_o, gamma, beta,
                                                      mmean, mvar, out);
}
